// MultiHeadAttention_64914135712222
// MI455X (gfx1250) — matrix operand order repaired and checked by reading, not compiled, not run on hardware
//
#include <hip/hip_runtime.h>

typedef __attribute__((ext_vector_type(16))) _Float16 v16h;
typedef __attribute__((ext_vector_type(8)))  _Float16 v8h;
typedef __attribute__((ext_vector_type(4)))  _Float16 v4h;
typedef __attribute__((ext_vector_type(8)))  float    v8f;

#define WMMA_F32_F16(a, b, c) \
  __builtin_amdgcn_wmma_f32_16x16x32_f16(false, (a), false, (b), (short)0, (c), false, false)

__device__ __forceinline__ v16h cat8(v8h lo, v8h hi) {
  return __builtin_shufflevector(lo, hi, 0,1,2,3,4,5,6,7,8,9,10,11,12,13,14,15);
}

// ---------------------------------------------------------------------------
// GEMM: Out = A[M,1024] @ W[1024,1024] + bias, via v_wmma_f32_16x16x32_f16.
// MODE 0: A fp32 (converted to f16), Out f16 scaled, head-major [B,H,S,64].
// MODE 1: A f16 (from ws), Out fp32 row-major [M,1024].
// Block: 256 threads (8 waves), tile 128(M) x 64(N), K-step 64 (8 WMMA/stage).
// ---------------------------------------------------------------------------
template <int MODE>
__global__ __launch_bounds__(256) void mha_gemm(const void* __restrict__ Ain,
                                                const float* __restrict__ W,
                                                const float* __restrict__ bias,
                                                void* __restrict__ Out,
                                                float scale) {
  constexpr int K = 1024, N = 1024;
  constexpr int BM = 128, BN = 64, BK = 64;
  constexpr int LDA = 72, LDB = 72;  // f16/row, 144B stride: 16B-aligned + conflict-free
  __shared__ __align__(16) _Float16 As[BM * LDA];
  __shared__ __align__(16) _Float16 Bt[BN * LDB];  // B tile stored transposed [n][k]

  const int tid = threadIdx.x;
  const int lane = tid & 31, wave = tid >> 5;
  const int wm = wave & 3, wn = wave >> 2;  // 4 waves along M, 2 along N
  const int m0 = blockIdx.y * BM, n0 = blockIdx.x * BN;

  v8f acc[2][2] = {};

  const int mA  = lane & 15;
  const int kbA = (lane < 16) ? 0 : 8;   // A layout: K 0-7/16-23 vs 8-15/24-31
  const int kbB = (lane < 16) ? 0 : 8;   // B layout: K 0-7/16-23 vs 8-15/24-31

  for (int k0 = 0; k0 < K; k0 += BK) {
    __syncthreads();
    // ---- stage A tile ----
    if (MODE == 0) {
      const float* A = (const float*)Ain;
#pragma unroll
      for (int i = 0; i < 8; ++i) {
        int id = tid + i * 256;  // 2048 float4 chunks (128 x 64 fp32)
        int row = id >> 4, c4 = id & 15;
        const float4 x = *(const float4*)&A[(size_t)(m0 + row) * K + k0 + c4 * 4];
        v4h hx;
        hx[0] = (_Float16)x.x; hx[1] = (_Float16)x.y;
        hx[2] = (_Float16)x.z; hx[3] = (_Float16)x.w;
        *(v4h*)&As[row * LDA + c4 * 4] = hx;
      }
    } else {
      const _Float16* A = (const _Float16*)Ain;
#pragma unroll
      for (int i = 0; i < 4; ++i) {
        int id = tid + i * 256;  // 1024 chunks of 8 f16
        int row = id >> 3, c8 = id & 7;
        *(v8h*)&As[row * LDA + c8 * 8] =
            *(const v8h*)&A[(size_t)(m0 + row) * K + k0 + c8 * 8];
      }
    }
    // ---- stage B tile transposed (fp32 -> f16) ----
#pragma unroll
    for (int i = 0; i < 4; ++i) {
      int id = tid + i * 256;  // 1024 float4 chunks (64 x 64 fp32)
      int row = id >> 4, c4 = id & 15;
      const float4 w = *(const float4*)&W[(size_t)(k0 + row) * N + n0 + c4 * 4];
      Bt[(c4 * 4 + 0) * LDB + row] = (_Float16)w.x;
      Bt[(c4 * 4 + 1) * LDB + row] = (_Float16)w.y;
      Bt[(c4 * 4 + 2) * LDB + row] = (_Float16)w.z;
      Bt[(c4 * 4 + 3) * LDB + row] = (_Float16)w.w;
    }
    if (k0 + BK < K)  // hint next W tile into cache (global_prefetch_b8)
      __builtin_prefetch(&W[(size_t)(k0 + BK + (tid >> 2)) * N + n0 + (tid & 3) * 16], 0, 1);
    __syncthreads();

    // ---- all fragments first (16x ds_load_b128), then 8 WMMAs ----
    v16h a[2][2], b[2][2];  // [ms|ns][ks]
#pragma unroll
    for (int ms = 0; ms < 2; ++ms) {
      const _Float16* p = &As[(wm * 32 + ms * 16 + mA) * LDA];
#pragma unroll
      for (int ks = 0; ks < 2; ++ks)
        a[ms][ks] = cat8(*(const v8h*)&p[ks * 32 + kbA],
                         *(const v8h*)&p[ks * 32 + kbA + 16]);
    }
#pragma unroll
    for (int ns = 0; ns < 2; ++ns) {
      const _Float16* p = &Bt[(wn * 32 + ns * 16 + mA) * LDB + kbB];
#pragma unroll
      for (int ks = 0; ks < 2; ++ks)
        b[ns][ks] = cat8(*(const v8h*)&p[ks * 32], *(const v8h*)&p[ks * 32 + 16]);
    }
#pragma unroll
    for (int ks = 0; ks < 2; ++ks)
#pragma unroll
      for (int ms = 0; ms < 2; ++ms)
#pragma unroll
        for (int ns = 0; ns < 2; ++ns)
          acc[ms][ns] = WMMA_F32_F16(a[ms][ks], b[ns][ks], acc[ms][ns]);
  }

  // ---- epilogue: bias, scale, store ----
  const int rofs = (lane < 16) ? 0 : 8;
#pragma unroll
  for (int ms = 0; ms < 2; ++ms)
#pragma unroll
    for (int ns = 0; ns < 2; ++ns) {
      const int gn = n0 + wn * 32 + ns * 16 + mA;
      const float bv = bias[gn];
#pragma unroll
      for (int r = 0; r < 8; ++r) {
        const int gm = m0 + wm * 32 + ms * 16 + r + rofs;
        const float val = (acc[ms][ns][r] + bv) * scale;
        if (MODE == 0) {
          _Float16* o = (_Float16*)Out;
          const int bb = gm >> 11, s = gm & 2047, hh = gn >> 6, d = gn & 63;
          o[(((size_t)bb * 16 + hh) * 2048 + s) * 64 + d] = (_Float16)val;
        } else {
          float* o = (float*)Out;
          o[(size_t)gm * 1024 + gn] = val;
        }
      }
    }
}

// ---------------------------------------------------------------------------
// Flash attention, base-2 online softmax (q pre-scaled by log2(e)/sqrt(dk)).
// 256 threads = 8 waves, 128 query rows per block (16 per wave); 64-key
// blocks staged in LDS (V transposed). 16 WMMAs per key block per wave.
// ---------------------------------------------------------------------------
__global__ __launch_bounds__(256) void mha_attn(const _Float16* __restrict__ q,
                                                const _Float16* __restrict__ k,
                                                const _Float16* __restrict__ v,
                                                _Float16* __restrict__ out) {
  constexpr int S = 2048, DK = 64;
  constexpr int LD = 72;  // 144B row stride everywhere
  __shared__ __align__(16) _Float16 Kb[64 * LD];      // K block [key][dk]
  __shared__ __align__(16) _Float16 Vt[64 * LD];      // V block transposed [d][key]
  __shared__ __align__(16) _Float16 Pb[8][16 * LD];   // per-wave P transpose buffer

  const int tid = threadIdx.x, lane = tid & 31, wave = tid >> 5;
  const int bh = blockIdx.y;                  // b*16 + h
  const int q0 = blockIdx.x * 128 + wave * 16;

  const _Float16* qbase = q + (size_t)bh * S * DK;
  const _Float16* kbase = k + (size_t)bh * S * DK;
  const _Float16* vbase = v + (size_t)bh * S * DK;

  const int mA  = lane & 15;
  const int kbA = (lane < 16) ? 0 : 8;
  const int kbB = (lane < 16) ? 0 : 8;

  // preload q A-fragments for this wave's 16 rows (dk 0-31 and 32-63)
  const _Float16* qrow = qbase + (size_t)(q0 + mA) * DK;
  const v16h qf0 = cat8(*(const v8h*)&qrow[kbA],      *(const v8h*)&qrow[kbA + 16]);
  const v16h qf1 = cat8(*(const v8h*)&qrow[32 + kbA], *(const v8h*)&qrow[32 + kbA + 16]);

  v8f o[4] = {};
  float m_s[8], l_s[8];
#pragma unroll
  for (int r = 0; r < 8; ++r) { m_s[r] = -3.0e38f; l_s[r] = 0.f; }

  _Float16* Pw = Pb[wave];

  for (int j0 = 0; j0 < S; j0 += 64) {
    __syncthreads();
    // cooperative stage of K (row-major) and V (transposed) 64-key blocks
#pragma unroll
    for (int i = 0; i < 2; ++i) {
      int id = tid + i * 256;  // 512 chunks of 8 f16
      int kr = id >> 3, c8 = id & 7;
      *(v8h*)&Kb[kr * LD + c8 * 8] = *(const v8h*)&kbase[(size_t)(j0 + kr) * DK + c8 * 8];
      v8h vv = *(const v8h*)&vbase[(size_t)(j0 + kr) * DK + c8 * 8];
#pragma unroll
      for (int j = 0; j < 8; ++j) Vt[(c8 * 8 + j) * LD + kr] = vv[j];
    }
    if (j0 + 64 < S) {
      __builtin_prefetch(&kbase[(size_t)(j0 + 64 + (tid >> 2)) * DK + (tid & 3) * 16], 0, 1);
      __builtin_prefetch(&vbase[(size_t)(j0 + 64 + (tid >> 2)) * DK + (tid & 3) * 16], 0, 1);
    }
    __syncthreads();

    // scores: four 16-key subtiles, s = q(16x64) @ k^T(64x16)
    // load all 16 B-fragment halves first, then 8 WMMAs (4 independent chains)
    v16h bl[4], bhh[4];
#pragma unroll
    for (int t = 0; t < 4; ++t) {
      const _Float16* kp = &Kb[(t * 16 + mA) * LD];
      bl[t]  = cat8(*(const v8h*)&kp[kbB],      *(const v8h*)&kp[kbB + 16]);
      bhh[t] = cat8(*(const v8h*)&kp[32 + kbB], *(const v8h*)&kp[32 + kbB + 16]);
    }
    v8f s[4] = {};
#pragma unroll
    for (int t = 0; t < 4; ++t) s[t] = WMMA_F32_F16(qf0, bl[t], s[t]);
#pragma unroll
    for (int t = 0; t < 4; ++t) s[t] = WMMA_F32_F16(qf1, bhh[t], s[t]);

    // base-2 online softmax; rows 0-7 live in lanes 0-15, rows 8-15 in
    // lanes 16-31, so xor masks 1/2/4/8 reduce exactly over one row.
    float p[4][8], alpha[8];
#pragma unroll
    for (int r = 0; r < 8; ++r) {
      float mx = fmaxf(fmaxf(s[0][r], s[1][r]), fmaxf(s[2][r], s[3][r]));
      mx = fmaxf(mx, __shfl_xor(mx, 1, 32));
      mx = fmaxf(mx, __shfl_xor(mx, 2, 32));
      mx = fmaxf(mx, __shfl_xor(mx, 4, 32));
      mx = fmaxf(mx, __shfl_xor(mx, 8, 32));
      const float mn = fmaxf(m_s[r], mx);
      alpha[r] = __builtin_amdgcn_exp2f(m_s[r] - mn);
      m_s[r] = mn;
      float rs = 0.f;
#pragma unroll
      for (int t = 0; t < 4; ++t) {
        p[t][r] = __builtin_amdgcn_exp2f(s[t][r] - mn);
        rs += p[t][r];
      }
      rs += __shfl_xor(rs, 1, 32);
      rs += __shfl_xor(rs, 2, 32);
      rs += __shfl_xor(rs, 4, 32);
      rs += __shfl_xor(rs, 8, 32);
      l_s[r] = l_s[r] * alpha[r] + rs;
    }
#pragma unroll
    for (int dt = 0; dt < 4; ++dt)
#pragma unroll
      for (int r = 0; r < 8; ++r) o[dt][r] *= alpha[r];

    // transpose P (C-layout -> A-layout) through per-wave LDS
    {
      const int row0 = (lane < 16) ? 0 : 8;
#pragma unroll
      for (int r = 0; r < 8; ++r)
#pragma unroll
        for (int t = 0; t < 4; ++t)
          Pw[(row0 + r) * LD + t * 16 + mA] = (_Float16)p[t][r];
    }
    asm volatile("s_wait_dscnt 0x0" ::: "memory");  // wave-local LDS visibility

    // o += P(16x64) @ V(64x64): load P + V fragments, then 8 WMMAs
    const _Float16* pp = &Pw[mA * LD];
    const v16h pf0 = cat8(*(const v8h*)&pp[kbA],      *(const v8h*)&pp[kbA + 16]);
    const v16h pf1 = cat8(*(const v8h*)&pp[32 + kbA], *(const v8h*)&pp[32 + kbA + 16]);
    v16h vf0[4], vf1[4];
#pragma unroll
    for (int dt = 0; dt < 4; ++dt) {
      const _Float16* vp = &Vt[(dt * 16 + mA) * LD + kbB];
      vf0[dt] = cat8(*(const v8h*)&vp[0],  *(const v8h*)&vp[16]);
      vf1[dt] = cat8(*(const v8h*)&vp[32], *(const v8h*)&vp[48]);
    }
#pragma unroll
    for (int dt = 0; dt < 4; ++dt) o[dt] = WMMA_F32_F16(pf0, vf0[dt], o[dt]);
#pragma unroll
    for (int dt = 0; dt < 4; ++dt) o[dt] = WMMA_F32_F16(pf1, vf1[dt], o[dt]);
  }

  // normalize and store f16 to [B, S, D_MODEL] for the output projection
  const int b = bh >> 4, h = bh & 15;
  const int row0 = (lane < 16) ? 0 : 8;
#pragma unroll
  for (int r = 0; r < 8; ++r) {
    const float inv = 1.0f / l_s[r];
    const int sr = q0 + row0 + r;
    _Float16* orow = out + ((size_t)b * S + sr) * 1024 + h * 64;
#pragma unroll
    for (int dt = 0; dt < 4; ++dt) orow[dt * 16 + mA] = (_Float16)(o[dt][r] * inv);
  }
}

// ---------------------------------------------------------------------------
extern "C" void kernel_launch(void* const* d_in, const int* in_sizes, int n_in,
                              void* d_out, int out_size, void* d_ws, size_t ws_size,
                              hipStream_t stream) {
  (void)in_sizes; (void)n_in; (void)out_size; (void)ws_size;
  const float* Q  = (const float*)d_in[0];
  const float* K  = (const float*)d_in[1];
  const float* V  = (const float*)d_in[2];
  const float* Wq = (const float*)d_in[3];
  const float* bq = (const float*)d_in[4];
  const float* Wk = (const float*)d_in[5];
  const float* bk = (const float*)d_in[6];
  const float* Wv = (const float*)d_in[7];
  const float* bv = (const float*)d_in[8];
  const float* Wo = (const float*)d_in[9];
  const float* bo = (const float*)d_in[10];

  constexpr size_t ELTS = (size_t)4096 * 1024;  // B*S*D_MODEL
  _Float16* qws = (_Float16*)d_ws;              // [B,H,S,64] f16
  _Float16* kws = qws + ELTS;
  _Float16* vws = kws + ELTS;
  _Float16* aws = vws + ELTS;                   // attn out [B,S,1024] f16

  const dim3 gemm_grid(16, 32);                 // N/64, M/128 (M = 4096)
  const dim3 gemm_blk(256);
  // q pre-scaled by log2(e)/sqrt(dk) so softmax can run in base 2
  const float qscale = 0.125f * 1.4426950408889634f;
  mha_gemm<0><<<gemm_grid, gemm_blk, 0, stream>>>(Q, Wq, bq, qws, qscale);
  mha_gemm<0><<<gemm_grid, gemm_blk, 0, stream>>>(K, Wk, bk, kws, 1.0f);
  mha_gemm<0><<<gemm_grid, gemm_blk, 0, stream>>>(V, Wv, bv, vws, 1.0f);

  mha_attn<<<dim3(16, 32), 256, 0, stream>>>(qws, kws, vws, aws);  // (S/128, B*H)

  mha_gemm<1><<<gemm_grid, gemm_blk, 0, stream>>>(aws, Wo, bo, d_out, 1.0f);
}
